// Siren_29025388987020
// MI455X (gfx1250) — compile-verified, matrix operand order checked by reading, not run on hardware
//
#include <hip/hip_runtime.h>
#include <stdint.h>

typedef __attribute__((ext_vector_type(16))) _Float16 v16h;
typedef __attribute__((ext_vector_type(8)))  float    v8f;
typedef __attribute__((ext_vector_type(2)))  __fp16   v2fp16;   // matches cvt_pkrtz return type

#define N_HID 256
#define HID   5
#define SF    4.774648292756860f   /* 30 / (2*pi): fold omega + sin-revolution scale into weights */

// pack two f32 -> one dword of two f16 (v_cvt_pk_rtz_f16_f32)
static __device__ __forceinline__ uint32_t pkh(float a, float b) {
    union { v2fp16 h; uint32_t u; } c;
    c.h = __builtin_amdgcn_cvt_pkrtz(a, b);
    return c.u;
}

union Frag { uint32_t u[8]; v16h v; };

// ---------------------------------------------------------------------------
// Setup: repack 256 hidden layers' f32 weights+bias into f16 A-fragment rows.
// table layout: [layer][row 0..7][4 dwords]  (rows 5..7 zero)
//   row m (< 5): d0 = pk(SF*w[m][0], SF*w[m][1])
//                d1 = pk(SF*w[m][2], SF*w[m][3])
//                d2 = pk(SF*w[m][4], SF*b[m])     <- bias folded in as K=5
//                d3 = 0
// ---------------------------------------------------------------------------
__global__ void siren_repack_kernel(const float* __restrict__ Wh,
                                    const float* __restrict__ bh,
                                    uint32_t* __restrict__ table)
{
    int idx = blockIdx.x * blockDim.x + threadIdx.x;   // 0 .. 256*8-1
    if (idx >= N_HID * 8) return;
    int l = idx >> 3, m = idx & 7;
    uint4 o; o.x = 0; o.y = 0; o.z = 0; o.w = 0;
    if (m < HID) {
        const float* w = Wh + (l * HID + m) * HID;
        float b = bh[l * HID + m];
        o.x = pkh(SF * w[0], SF * w[1]);
        o.y = pkh(SF * w[2], SF * w[3]);
        o.z = pkh(SF * w[4], SF * b);
    }
    reinterpret_cast<uint4*>(table)[idx] = o;
}

// ---------------------------------------------------------------------------
// Main kernel. Block = 256 threads = 8 waves. Each wave handles 96 points as
// three independent 32-point WMMA chains (B0/B1/B2) sharing the per-layer A.
//
// One WMMA carries 2 point-tiles: tile-lo (lanes 0-15) -> D rows 0-4,
// tile-hi (lanes 16-31) -> D rows 8-12. A holds W in rows 0-4 @K0-5 and in
// rows 8-12 duplicated at BOTH K8-13 (lanes 24-28, v0-2) and K16-21
// (lanes 8-12, v4-6) so the mapping is correct under either legal B layout;
// the unused duplicate multiplies guaranteed-zero B slots.
// Recurrence is lane-local: d[m] -> v_sin -> v_cvt_pk -> next B, all 32 lanes.
// ---------------------------------------------------------------------------
__global__ __launch_bounds__(256) void siren_kernel(
    const float* __restrict__ coords,
    const float* __restrict__ Wf, const float* __restrict__ bf,     // first layer
    const uint32_t* __restrict__ tableg,                            // repacked hidden
    const float* __restrict__ Wo, const float* __restrict__ bo,     // final layer
    float* __restrict__ out, int N)
{
    // 32 KB A-fragment table + 16 B zero slot
    __shared__ uint32_t lds[N_HID * 8 * 4 + 4];

    for (int i = threadIdx.x; i < N_HID * 8; i += 256)
        reinterpret_cast<uint4*>(lds)[i] = reinterpret_cast<const uint4*>(tableg)[i];
    if (threadIdx.x == 0) {
        lds[N_HID * 32 + 0] = 0; lds[N_HID * 32 + 1] = 0;
        lds[N_HID * 32 + 2] = 0; lds[N_HID * 32 + 3] = 0;
    }
    __syncthreads();

    const int lane = threadIdx.x & 31;
    const int wave = threadIdx.x >> 5;
    const int base = (blockIdx.x * 8 + wave) * 96;       // 96 points per wave
    int pA = base + lane;       if (pA >= N) pA = N - 1;
    int pB = base + 32 + lane;  if (pB >= N) pB = N - 1;
    int pC = base + 64 + lane;  if (pC >= N) pC = N - 1;

    // ---- first layer (f32 VALU, scalar-uniform weights), all 32 lanes ----
    float a0 = coords[3 * pA], a1 = coords[3 * pA + 1], a2 = coords[3 * pA + 2];
    float b0 = coords[3 * pB], b1 = coords[3 * pB + 1], b2 = coords[3 * pB + 2];
    float c0 = coords[3 * pC], c1 = coords[3 * pC + 1], c2 = coords[3 * pC + 2];
    float sa[HID], sb[HID], sc[HID];
#pragma unroll
    for (int m = 0; m < HID; ++m) {
        float w0 = Wf[3 * m], w1 = Wf[3 * m + 1], w2 = Wf[3 * m + 2], bb = bf[m];
        sa[m] = __builtin_amdgcn_sinf(SF * (a0 * w0 + a1 * w1 + a2 * w2 + bb));
        sb[m] = __builtin_amdgcn_sinf(SF * (b0 * w0 + b1 * w1 + b2 * w2 + bb));
        sc[m] = __builtin_amdgcn_sinf(SF * (c0 * w0 + c1 * w1 + c2 * w2 + bb));
    }

    // Persistent fragments: zero once, only low dwords rewritten in the loop.
    Frag A, B0, B1, B2;
#pragma unroll
    for (int i = 0; i < 8; ++i) { A.u[i] = 0; B0.u[i] = 0; B1.u[i] = 0; B2.u[i] = 0; }
    B0.u[0] = pkh(sa[0], sa[1]); B0.u[1] = pkh(sa[2], sa[3]); B0.u[2] = pkh(sa[4], 1.0f);
    B1.u[0] = pkh(sb[0], sb[1]); B1.u[1] = pkh(sb[2], sb[3]); B1.u[2] = pkh(sb[4], 1.0f);
    B2.u[0] = pkh(sc[0], sc[1]); B2.u[1] = pkh(sc[2], sc[3]); B2.u[2] = pkh(sc[4], 1.0f);

    // A-fragment addresses:
    //  v0-3: lanes 0-7 (rows 0-7, K0-7) and lanes 24-31 (rows 8-15, K8-15) read
    //        the weight row (lane&7); all other lanes read zeros.
    //  v4-7: lanes 8-15 (rows 8-15, K16-23) read weight row (lane&7); else zeros.
    const uint32_t Z = (uint32_t)(N_HID * 128);
    const bool use0 = (lane < 8) || (lane >= 24);
    const bool use1 = (lane >= 8) && (lane < 16);
    uint32_t addr0 = use0 ? (uint32_t)((lane & 7) * 16) : Z;
    uint32_t addr1 = use1 ? (uint32_t)((lane & 7) * 16) : Z;
    uint32_t str0  = use0 ? 128u : 0u;
    uint32_t str1  = use1 ? 128u : 0u;
    const char* lbase = reinterpret_cast<const char*>(lds);

    v8f zeroC = {};
    float xa[HID], xb[HID], xc[HID];
#pragma unroll
    for (int m = 0; m < HID; ++m) { xa[m] = 0.0f; xb[m] = 0.0f; xc[m] = 0.0f; }

#pragma unroll 2
    for (int l = 0; l < N_HID; ++l) {
        uint4 lo = *reinterpret_cast<const uint4*>(lbase + addr0);  // ds_load_b128
        uint4 hi = *reinterpret_cast<const uint4*>(lbase + addr1);  // ds_load_b128
        addr0 += str0; addr1 += str1;
        A.u[0] = lo.x; A.u[1] = lo.y; A.u[2] = lo.z; A.u[3] = lo.w;
        A.u[4] = hi.x; A.u[5] = hi.y; A.u[6] = hi.z; A.u[7] = hi.w;

        v8f d0 = __builtin_amdgcn_wmma_f32_16x16x32_f16(
            false, A.v, false, B0.v, (short)0, zeroC, false, false);
        v8f d1 = __builtin_amdgcn_wmma_f32_16x16x32_f16(
            false, A.v, false, B1.v, (short)0, zeroC, false, false);
        v8f d2 = __builtin_amdgcn_wmma_f32_16x16x32_f16(
            false, A.v, false, B2.v, (short)0, zeroC, false, false);

#pragma unroll
        for (int m = 0; m < HID; ++m) {
            xa[m] = __builtin_amdgcn_sinf(d0[m]);   // lanes 0-15: tile-lo, 16-31: tile-hi
            xb[m] = __builtin_amdgcn_sinf(d1[m]);
            xc[m] = __builtin_amdgcn_sinf(d2[m]);
        }
        B0.u[0] = pkh(xa[0], xa[1]); B0.u[1] = pkh(xa[2], xa[3]); B0.u[2] = pkh(xa[4], 1.0f);
        B1.u[0] = pkh(xb[0], xb[1]); B1.u[1] = pkh(xb[2], xb[3]); B1.u[2] = pkh(xb[4], 1.0f);
        B2.u[0] = pkh(xc[0], xc[1]); B2.u[1] = pkh(xc[2], xc[3]); B2.u[2] = pkh(xc[4], 1.0f);
    }

    // ---- final linear layer (1 output), f32 VALU, all 32 lanes ----
    {
        float w0 = Wo[0], w1 = Wo[1], w2 = Wo[2], w3 = Wo[3], w4 = Wo[4], bb = bo[0];
        float o0 = xa[0] * w0 + xa[1] * w1 + xa[2] * w2 + xa[3] * w3 + xa[4] * w4 + bb;
        float o1 = xb[0] * w0 + xb[1] * w1 + xb[2] * w2 + xb[3] * w3 + xb[4] * w4 + bb;
        float o2 = xc[0] * w0 + xc[1] * w1 + xc[2] * w2 + xc[3] * w3 + xc[4] * w4 + bb;
        int q0 = base + lane, q1 = base + 32 + lane, q2 = base + 64 + lane;
        if (q0 < N) out[q0] = o0;
        if (q1 < N) out[q1] = o1;
        if (q2 < N) out[q2] = o2;
    }
}

// ---------------------------------------------------------------------------
extern "C" void kernel_launch(void* const* d_in, const int* in_sizes, int n_in,
                              void* d_out, int out_size, void* d_ws, size_t ws_size,
                              hipStream_t stream)
{
    const float* coords   = (const float*)d_in[0];   // [N,3]
    const float* W_first  = (const float*)d_in[1];   // [5,3]
    const float* b_first  = (const float*)d_in[2];   // [5]
    const float* W_hidden = (const float*)d_in[3];   // [256,5,5]
    const float* b_hidden = (const float*)d_in[4];   // [256,5]
    const float* W_final  = (const float*)d_in[5];   // [1,5]
    const float* b_final  = (const float*)d_in[6];   // [1]
    float* out = (float*)d_out;
    int N = in_sizes[0] / 3;

    uint32_t* table = (uint32_t*)d_ws;               // 32 KB repacked weights

    siren_repack_kernel<<<(N_HID * 8 + 255) / 256, 256, 0, stream>>>(
        W_hidden, b_hidden, table);

    int blocks = (N + 767) / 768;                    // 768 points per block
    siren_kernel<<<blocks, 256, 0, stream>>>(
        coords, W_first, b_first, table, W_final, b_final, out, N);

    // tuple output: (out [N,1], coords [N,3]) concatenated flat
    (void)hipMemcpyAsync(out + N, coords, (size_t)3 * N * sizeof(float),
                         hipMemcpyDeviceToDevice, stream);
}
